// CausalSelfAttentionAdaLN_50294067036697
// MI455X (gfx1250) — matrix operand order repaired and checked by reading, not compiled, not run on hardware
//
#include <hip/hip_runtime.h>
#include <hip/hip_bf16.h>

// ---------------------------------------------------------------------------
// CausalSelfAttentionAdaLN for MI455X (gfx1250): all matmuls via
// v_wmma_f32_16x16x32_bf16, softmax/reductions in fp32 VALU, flash-attention
// streaming with K tiles staged via GLOBAL_LOAD_ASYNC_TO_LDS (ASYNCcnt) and
// V tiles staged transposed through VGPRs.
// ---------------------------------------------------------------------------

#define BATCH 2
#define SEQ   2048
#define DMODEL 2048
#define NHEADS 16
#define HEADD  128
#define MROWS  (BATCH * SEQ)        // 4096
#define QKVN   (3 * DMODEL)         // 6144

typedef __attribute__((ext_vector_type(16))) __bf16 v16bf;
typedef __attribute__((ext_vector_type(8)))  float  v8f;

union Frag16 { v16bf v; unsigned u[8]; unsigned short h[16]; };
union FragC  { v8f   v; float f[8]; };

__device__ __forceinline__ __bf16 f2bf(float x) {
  union { float f; unsigned u; } a; a.f = x;
  unsigned r = a.u + 0x7FFFu + ((a.u >> 16) & 1u);   // round-to-nearest-even
  union { unsigned short s; __bf16 b; } o; o.s = (unsigned short)(r >> 16);
  return o.b;
}

__device__ __forceinline__ v8f wmma_bf16(v16bf a, v16bf b, v8f c) {
  // D = A(16x32 bf16) * B(32x16 bf16) + C(16x16 f32)
  return __builtin_amdgcn_wmma_f32_16x16x32_bf16(
      /*neg_a=*/false, a, /*neg_b=*/false, b,
      /*c_mod=*/(short)0, c, /*reuse_a=*/false, /*reuse_b=*/false);
}

// Wave-relative LDS byte offset of a generic pointer to __shared__ memory
// (addrspacecast(3->flat) keeps the LDS offset in the low 32 bits).
__device__ __forceinline__ unsigned lds_off(const void* p) {
  return (unsigned)(unsigned long long)p;
}

// ---------------------------------------------------------------------------
// Kernel 1: AdaLN (per-row LayerNorm with tanh-gated gamma), fp32 -> bf16
// One block of 256 threads per row (D=2048 -> 8 elements/thread).
// ---------------------------------------------------------------------------
__global__ void adaln_kernel(const float* __restrict__ x,
                             const float* __restrict__ gamma,
                             const float* __restrict__ beta,
                             __bf16* __restrict__ xn) {
  const int row = blockIdx.x;          // 0..4095
  const int tid = threadIdx.x;         // 0..255
  const size_t base = (size_t)row * DMODEL;

  float vals[8];
  float s = 0.f, s2 = 0.f;
#pragma unroll
  for (int i = 0; i < 8; ++i) {
    float v = x[base + tid + i * 256];
    vals[i] = v; s += v; s2 += v * v;
  }
#pragma unroll
  for (int off = 16; off > 0; off >>= 1) {
    s  += __shfl_xor(s,  off, 32);
    s2 += __shfl_xor(s2, off, 32);
  }
  __shared__ float red[2][8];
  const int wave = tid >> 5;
  if ((tid & 31) == 0) { red[0][wave] = s; red[1][wave] = s2; }
  __syncthreads();
  float ts = 0.f, ts2 = 0.f;
#pragma unroll
  for (int i = 0; i < 8; ++i) { ts += red[0][i]; ts2 += red[1][i]; }
  const float inv_d = 1.0f / (float)DMODEL;
  const float mean = ts * inv_d;
  const float var  = ts2 * inv_d - mean * mean;
  const float rstd = rsqrtf(var + 1e-6f);
#pragma unroll
  for (int i = 0; i < 8; ++i) {
    const int c = tid + i * 256;
    const float g = 1.0f + 0.1f * tanhf((gamma[base + c] - 1.0f) * 10.0f);
    const float y = g * ((vals[i] - mean) * rstd) + beta[base + c];
    xn[base + c] = f2bf(y);
  }
}

// ---------------------------------------------------------------------------
// Kernel 2: weight conversion fp32 -> bf16 (w_qkv then w_proj)
// ---------------------------------------------------------------------------
__global__ void cvt_weights_kernel(const float* __restrict__ wqkv,
                                   const float* __restrict__ wproj,
                                   __bf16* __restrict__ wq,
                                   __bf16* __restrict__ wp) {
  const size_t NQ = (size_t)QKVN * DMODEL;     // 12,582,912
  const size_t NP = (size_t)DMODEL * DMODEL;   //  4,194,304
  size_t i = (size_t)blockIdx.x * blockDim.x + threadIdx.x;
  const size_t stride = (size_t)gridDim.x * blockDim.x;
  for (size_t k = i; k < NQ + NP; k += stride) {
    if (k < NQ) wq[k] = f2bf(wqkv[k]);
    else        wp[k - NQ] = f2bf(wproj[k - NQ]);
  }
}

// ---------------------------------------------------------------------------
// Kernel 3/5: GEMM  C[m][n] = sum_k A[m][k] * W[n][k]   (A: MxK, W: NxK, bf16)
// Block = 256 threads = 8 waves; block tile 128x128; wave tile 32x64
// (2 M-tiles x 4 N-tiles of 16x16 WMMA fragments). K stepped by 32.
// mode 0: store fp32 dense (M x N) to outF.
// mode 1: scatter bf16 into Q/K/V buffers laid out (B,H,S,Hd).
// ---------------------------------------------------------------------------
__global__ void gemm_bf16_kernel(const __bf16* __restrict__ A,
                                 const __bf16* __restrict__ W,
                                 int M, int N, int K, int mode,
                                 float* __restrict__ outF,
                                 __bf16* __restrict__ outQ,
                                 __bf16* __restrict__ outK,
                                 __bf16* __restrict__ outV) {
  const int lane = threadIdx.x & 31;
  const int wave = threadIdx.x >> 5;
  const int rowl = lane & 15;   // A row / B column within a fragment
  const int hi   = lane >> 4;   // lane-half

  const int m0 = blockIdx.y * 128 + (wave >> 1) * 32;
  const int n0 = blockIdx.x * 128 + (wave & 1) * 64;

  // A-fragment K offsets per VGPR pair (contiguous order k = 16*hi + s, matches B)
  int koffA[8];
#pragma unroll
  for (int j = 0; j < 8; ++j)
    koffA[j] = 2 * j + 16 * hi;

  FragC c[2][4];
#pragma unroll
  for (int mt = 0; mt < 2; ++mt)
#pragma unroll
    for (int nt = 0; nt < 4; ++nt)
#pragma unroll
      for (int r = 0; r < 8; ++r) c[mt][nt].f[r] = 0.f;

  for (int k0 = 0; k0 < K; k0 += 32) {
    Frag16 a[2];
#pragma unroll
    for (int mt = 0; mt < 2; ++mt) {
      const __bf16* ap = A + (size_t)(m0 + mt * 16 + rowl) * K + k0;
#pragma unroll
      for (int j = 0; j < 8; ++j)
        a[mt].u[j] = *(const unsigned*)(ap + koffA[j]);
    }
    Frag16 b[4];
#pragma unroll
    for (int nt = 0; nt < 4; ++nt) {
      // B[k][n] = W[n0+nt*16+n][k0+k]; lane holds column n = rowl, K-half = hi
      const __bf16* wp = W + (size_t)(n0 + nt * 16 + rowl) * K + k0 + 16 * hi;
#pragma unroll
      for (int j = 0; j < 8; ++j)
        b[nt].u[j] = *(const unsigned*)(wp + 2 * j);
    }
#pragma unroll
    for (int mt = 0; mt < 2; ++mt)
#pragma unroll
      for (int nt = 0; nt < 4; ++nt)
        c[mt][nt].v = wmma_bf16(a[mt].v, b[nt].v, c[mt][nt].v);
  }

  // Epilogue: C-fragment element (r) is at M = base + r + 8*hi, N = base + rowl
#pragma unroll
  for (int mt = 0; mt < 2; ++mt) {
#pragma unroll
    for (int nt = 0; nt < 4; ++nt) {
#pragma unroll
      for (int r = 0; r < 8; ++r) {
        const int m = m0 + mt * 16 + r + 8 * hi;
        const int n = n0 + nt * 16 + rowl;
        const float v = c[mt][nt].f[r];
        if (mode == 0) {
          outF[(size_t)m * N + n] = v;
        } else {
          const int bb = m >> 11;            // m / 2048
          const int s  = m & 2047;
          const int which = n >> 11;         // 0=q 1=k 2=v
          const int hh = (n >> 7) & 15;
          const int d  = n & 127;
          const size_t idx =
              (((size_t)(bb * NHEADS + hh) * SEQ) + s) * HEADD + d;
          const __bf16 bv = f2bf(v);
          if (which == 0) outQ[idx] = bv;
          else if (which == 1) outK[idx] = bv;
          else outV[idx] = bv;
        }
      }
    }
  }
}

// ---------------------------------------------------------------------------
// Kernel 4: causal flash attention.
// One block per (b, h, 128-row q slab). 8 waves, each owns 16 q rows:
//   Q resident as 4 A-fragments, O as 8 fp32 C-fragments (16x128).
// Per 32-key step: K tile DMA'd to LDS with GLOBAL_LOAD_ASYNC_TO_LDS_B128
// (ASYNCcnt), V tile staged transposed through VGPRs, S = QK^T via 8 WMMAs,
// online softmax with 16-lane shfl reductions, P staged through LDS
// (C-frag -> A-frag relayout), O += P*V via 8 WMMAs.
// ---------------------------------------------------------------------------
__global__ void attn_kernel(const __bf16* __restrict__ qb,
                            const __bf16* __restrict__ kb,
                            const __bf16* __restrict__ vb,
                            __bf16* __restrict__ ab) {
  const int lane = threadIdx.x & 31;
  const int wave = threadIdx.x >> 5;
  const int rowl = lane & 15;
  const int hi   = lane >> 4;

  const int tile = blockIdx.x;        // 0..511
  const int qblk = tile & 15;         // S/128
  const int hh   = (tile >> 4) & 15;
  const int bb   = tile >> 8;
  const int q0   = qblk * 128 + wave * 16;
  const size_t base = (size_t)(bb * NHEADS + hh) * SEQ * HEADD;

  __shared__ __bf16 lK[32][HEADD];        // K tile, row-major         (8 KB)
  __shared__ __bf16 lV[HEADD][32];        // V tile, transposed [d][k] (8 KB)
  __shared__ __bf16 lP[8][16][32];        // per-wave P tile           (8 KB)

  int koffA[8];
#pragma unroll
  for (int j = 0; j < 8; ++j)
    koffA[j] = 2 * j + 16 * hi;

  // Resident Q fragments (16 rows x 128 cols -> 4 A-frags over Hd)
  Frag16 qf[4];
  {
    const __bf16* qp = qb + base + (size_t)(q0 + rowl) * HEADD;
#pragma unroll
    for (int t = 0; t < 4; ++t)
#pragma unroll
      for (int j = 0; j < 8; ++j)
        qf[t].u[j] = *(const unsigned*)(qp + t * 32 + koffA[j]);
  }

  FragC o[8];
#pragma unroll
  for (int t = 0; t < 8; ++t)
#pragma unroll
    for (int r = 0; r < 8; ++r) o[t].f[r] = 0.f;
  float mrow[8], lrow[8];
#pragma unroll
  for (int r = 0; r < 8; ++r) { mrow[r] = -1e30f; lrow[r] = 0.f; }

  const float scale = 0.08838834764831845f;   // 1/sqrt(128)
  const int jend = qblk * 128 + 127;          // block-wide causal bound

  for (int j0 = 0; j0 <= jend; j0 += 32) {
    __syncthreads();   // previous iteration finished reading lK/lV
    {
      // Cooperative staging: 256 threads cover 32 rows x 8 col-groups of 16.
      const int t  = threadIdx.x;
      const int kr = t >> 3;            // 0..31 key row within tile
      const int cg = (t & 7) * 16;      // 0..112 column group
      const int src = (j0 + kr < SEQ - 1) ? (j0 + kr) : (SEQ - 1);

      // --- K tile: async DMA directly into LDS (no VGPR round trip) ---
      {
        const unsigned long long gk =
            (unsigned long long)(kb + base + (size_t)src * HEADD + cg);
        const unsigned lk = lds_off(&lK[kr][cg]);
        asm volatile("global_load_async_to_lds_b128 %0, %1, off"
                     :: "v"(lk), "v"(gk) : "memory");
        asm volatile("global_load_async_to_lds_b128 %0, %1, off offset:16"
                     :: "v"(lk), "v"(gk) : "memory");
      }

      // --- V tile: through VGPRs with transpose ([d][k] layout in LDS) ---
      const __bf16* vp = vb + base + (size_t)src * HEADD + cg;
      __align__(16) __bf16 tmp[16];
      *(uint4*)&tmp[0] = *(const uint4*)(vp);
      *(uint4*)&tmp[8] = *(const uint4*)(vp + 8);
#pragma unroll
      for (int i = 0; i < 16; ++i) lV[cg + i][kr] = tmp[i];

      // Prefetch next tile while this one is consumed.
      if (j0 + 32 <= jend) {
        const int nsrc = (j0 + 32 + kr < SEQ - 1) ? (j0 + 32 + kr) : (SEQ - 1);
        __builtin_prefetch(vb + base + (size_t)nsrc * HEADD + cg, 0, 1);
      }
      // Async K copy must have landed in LDS before the block barrier.
      asm volatile("s_wait_asynccnt 0x0" ::: "memory");
    }
    __syncthreads();

    // S = Q * K^T : two 16x16 column tiles, K-dim (Hd) in 4 steps of 32
    FragC sfr[2];
#pragma unroll
    for (int f = 0; f < 2; ++f) {
#pragma unroll
      for (int r = 0; r < 8; ++r) sfr[f].f[r] = 0.f;
#pragma unroll
      for (int t = 0; t < 4; ++t) {
        Frag16 bf_;
        // B[k][n] = K[j0 + f*16 + n][t*32 + k]; lane column n = rowl
        const __bf16* ks = &lK[f * 16 + rowl][t * 32 + 16 * hi];
#pragma unroll
        for (int j = 0; j < 8; ++j)
          bf_.u[j] = *(const unsigned*)(ks + 2 * j);
        sfr[f].v = wmma_bf16(qf[t].v, bf_.v, sfr[f].v);
      }
    }

    // Online softmax (row = q index; a row lives across one 16-lane half)
#pragma unroll
    for (int r = 0; r < 8; ++r) {
      const int qi  = q0 + r + 8 * hi;
      const int k0c = j0 + rowl;
      const int k1c = j0 + 16 + rowl;
      float v0 = (k0c <= qi) ? sfr[0].f[r] * scale : -1e30f;
      float v1 = (k1c <= qi) ? sfr[1].f[r] * scale : -1e30f;
      float mx = fmaxf(v0, v1);
#pragma unroll
      for (int off = 1; off < 16; off <<= 1)
        mx = fmaxf(mx, __shfl_xor(mx, off, 32));
      const float mnew = fmaxf(mrow[r], mx);
      const float corr = __expf(mrow[r] - mnew);
      const float p0 = (k0c <= qi) ? __expf(v0 - mnew) : 0.f;
      const float p1 = (k1c <= qi) ? __expf(v1 - mnew) : 0.f;
      float ps = p0 + p1;
#pragma unroll
      for (int off = 1; off < 16; off <<= 1)
        ps += __shfl_xor(ps, off, 32);
      lrow[r] = lrow[r] * corr + ps;
      mrow[r] = mnew;
#pragma unroll
      for (int t = 0; t < 8; ++t) o[t].f[r] *= corr;
      lP[wave][r + 8 * hi][rowl]      = f2bf(p0);
      lP[wave][r + 8 * hi][16 + rowl] = f2bf(p1);
    }

    // Wave-private LDS round-trip: DS ops are in-order per wave.
    asm volatile("s_wait_dscnt 0" ::: "memory");

    // O += P * V : P as A-fragment (16x32), V columns from transposed LDS
    Frag16 pf;
    {
      const __bf16* ps = &lP[wave][rowl][0];
#pragma unroll
      for (int j = 0; j < 8; ++j)
        pf.u[j] = *(const unsigned*)(ps + koffA[j]);
    }
#pragma unroll
    for (int t = 0; t < 8; ++t) {
      Frag16 bv_;
      // B[k][n] = V[j0+k][t*16+n] = lV[t*16+n][k]; pairs contiguous in k
      const __bf16* vs = &lV[t * 16 + rowl][16 * hi];
#pragma unroll
      for (int j = 0; j < 8; ++j)
        bv_.u[j] = *(const unsigned*)(vs + 2 * j);
      o[t].v = wmma_bf16(pf.v, bv_.v, o[t].v);
    }
  }

  // Normalize and store to (B,S,D) bf16 for the projection GEMM.
#pragma unroll
  for (int r = 0; r < 8; ++r) {
    const float inv = 1.0f / lrow[r];
    const int s = q0 + r + 8 * hi;
    const size_t ob = ((size_t)(bb * SEQ + s)) * DMODEL + hh * HEADD;
#pragma unroll
    for (int t = 0; t < 8; ++t)
      ab[ob + t * 16 + rowl] = f2bf(o[t].f[r] * inv);
  }
}

// ---------------------------------------------------------------------------
// Host-side launcher.
// ---------------------------------------------------------------------------
extern "C" void kernel_launch(void* const* d_in, const int* in_sizes, int n_in,
                              void* d_out, int out_size, void* d_ws, size_t ws_size,
                              hipStream_t stream) {
  (void)in_sizes; (void)n_in; (void)out_size; (void)ws_size;
  const float* x     = (const float*)d_in[0];
  const float* gamma = (const float*)d_in[1];
  const float* beta  = (const float*)d_in[2];
  const float* wqkv  = (const float*)d_in[3];
  const float* wproj = (const float*)d_in[4];
  float* out = (float*)d_out;

  // Workspace layout (bf16 elements):
  const size_t XN_E = (size_t)MROWS * DMODEL;     //  8,388,608
  const size_t WQ_E = (size_t)QKVN * DMODEL;      // 12,582,912
  const size_t WP_E = (size_t)DMODEL * DMODEL;    //  4,194,304
  const size_t QB_E = (size_t)BATCH * NHEADS * SEQ * HEADD;  // 8,388,608
  __bf16* xn  = (__bf16*)d_ws;
  __bf16* wq  = xn + XN_E;
  __bf16* wp  = wq + WQ_E;
  __bf16* qb  = wp + WP_E;
  __bf16* kb  = qb + QB_E;
  __bf16* vb  = kb + QB_E;
  __bf16* ab  = vb + QB_E;   // total ~117 MB

  adaln_kernel<<<MROWS, 256, 0, stream>>>(x, gamma, beta, xn);
  cvt_weights_kernel<<<8192, 256, 0, stream>>>(wqkv, wproj, wq, wp);
  // QKV GEMM: M=4096, N=6144, K=2048 -> grid (N/128, M/128)
  gemm_bf16_kernel<<<dim3(QKVN / 128, MROWS / 128), 256, 0, stream>>>(
      xn, wq, MROWS, QKVN, DMODEL, /*mode=*/1, nullptr, qb, kb, vb);
  // Attention: one block per (b, h, 128-row slab)
  attn_kernel<<<BATCH * NHEADS * (SEQ / 128), 256, 0, stream>>>(qb, kb, vb, ab);
  // Projection GEMM: M=4096, N=2048, K=2048 -> fp32 output
  gemm_bf16_kernel<<<dim3(DMODEL / 128, MROWS / 128), 256, 0, stream>>>(
      ab, wp, MROWS, DMODEL, DMODEL, /*mode=*/0, out, nullptr, nullptr, nullptr);
}
